// GAT_Model_35785667510798
// MI455X (gfx1250) — matrix operand order repaired and checked by reading, not compiled, not run on hardware
//
#include <hip/hip_runtime.h>
#include <hip/hip_bf16.h>
#include <math.h>

// ---------------------------------------------------------------------------
// GAT forward for MI455X (gfx1250, wave32, WMMA).
// Layers: h = relu( GATConv(x) ) x3 ; head-mean ; per-graph max/mean/sum pool;
// final linear. GEMMs use v_wmma_f32_16x16x32_f16 (f16 in, f32 accum), with
// 4x N-dimension register blocking so each A load feeds 4 WMMAs.
// ---------------------------------------------------------------------------

#define N_NODES 50000
#define N_EDGES 800000
#define N_EE    (N_EDGES + N_NODES)   // with self loops
#define N_B     512
#define F_IN    128
#define HC      256                    // H * C
#define H_HEADS 4
#define C_DIM   64
#define N_OUT   10
#define NEG_SLOPE 0.2f

typedef __attribute__((ext_vector_type(16))) _Float16 v16h;
typedef __attribute__((ext_vector_type(8)))  _Float16 v8h;
typedef __attribute__((ext_vector_type(8)))  float    v8f;

// ---------------------------- small utilities ------------------------------

__global__ void fill_f32_kernel(float* __restrict__ p, float v, int n) {
    int i = blockIdx.x * blockDim.x + threadIdx.x;
    if (i < n) p[i] = v;
}

__global__ void cvt_f32_to_f16_kernel(const float* __restrict__ src,
                                      _Float16* __restrict__ dst, int n) {
    int i = blockIdx.x * blockDim.x + threadIdx.x;
    if (i < n) dst[i] = (_Float16)src[i];
}

// W is [K, 256] row-major (f32). Produce Wt16 [256, K] (f16) so that a WMMA
// B-tile column is a contiguous run of K values.
__global__ void transpose_cvt_w_kernel(const float* __restrict__ W,
                                       _Float16* __restrict__ Wt, int K) {
    int i = blockIdx.x * blockDim.x + threadIdx.x;   // i = k*256 + c
    if (i < K * HC) {
        int k = i / HC;
        int c = i % HC;
        Wt[(size_t)c * K + k] = (_Float16)W[i];
    }
}

__device__ __forceinline__ void atomicMaxF(float* addr, float v) {
    // signed-int max for non-negatives, unsigned-int min for negatives;
    // correct for buffers initialized to -inf.
    if (v >= 0.0f) atomicMax((int*)addr, __float_as_int(v));
    else           atomicMin((unsigned int*)addr, (unsigned int)__float_as_int(v));
}

// ------------------------------ WMMA GEMM ----------------------------------
// C[N,256] = A[N,K](f16) x Wt[256,K](f16, pre-transposed B)
// One wave per 16x64 output strip: 4 accumulators (N-blocking) share each
// A-fragment load, so A traffic drops 4x and WMMA density per load rises 4x.
__global__ __launch_bounds__(32)
void wmma_gemm_kernel(const _Float16* __restrict__ A,
                      const _Float16* __restrict__ Bt,
                      float* __restrict__ Cout, int K) {
    const int tile = blockIdx.x;
    const int tm  = tile >> 2;          // 256/64 = 4 strips of 4 col tiles
    const int tn0 = (tile & 3) * 4;     // first 16-wide col tile of the strip
    const int lane = threadIdx.x;
    const int lmod = lane & 15;
    const int lhi  = lane >> 4;         // 0 or 1

    // A lane layout (16-bit A 16x32): M = lane%16; elems 0-15 -> K = base+lhi*16 .. +15,
    // matching B's contiguous order.
    const _Float16* aptr = A + (size_t)(tm * 16 + lmod) * K + lhi * 16;
    // B lane layout (16-bit B 32x16): N = lane%16; K = base + lhi*16 .. +15,
    // contiguous thanks to the transposed weight buffer.
    const _Float16* bptr0 = Bt + (size_t)((tn0 + 0) * 16 + lmod) * K + lhi * 16;
    const _Float16* bptr1 = Bt + (size_t)((tn0 + 1) * 16 + lmod) * K + lhi * 16;
    const _Float16* bptr2 = Bt + (size_t)((tn0 + 2) * 16 + lmod) * K + lhi * 16;
    const _Float16* bptr3 = Bt + (size_t)((tn0 + 3) * 16 + lmod) * K + lhi * 16;

    v8f acc0 = {}, acc1 = {}, acc2 = {}, acc3 = {};
    for (int kk = 0; kk < K; kk += 32) {
        v8h alo = *(const v8h*)(aptr + kk);
        v8h ahi = *(const v8h*)(aptr + kk + 8);
        v16h av = __builtin_shufflevector(alo, ahi,
                   0,1,2,3,4,5,6,7,8,9,10,11,12,13,14,15);
        v16h bv0 = *(const v16h*)(bptr0 + kk);
        v16h bv1 = *(const v16h*)(bptr1 + kk);
        v16h bv2 = *(const v16h*)(bptr2 + kk);
        v16h bv3 = *(const v16h*)(bptr3 + kk);
        acc0 = __builtin_amdgcn_wmma_f32_16x16x32_f16(
                   false, av, false, bv0, (short)0, acc0, false, false);
        acc1 = __builtin_amdgcn_wmma_f32_16x16x32_f16(
                   false, av, false, bv1, (short)0, acc1, false, false);
        acc2 = __builtin_amdgcn_wmma_f32_16x16x32_f16(
                   false, av, false, bv2, (short)0, acc2, false, false);
        acc3 = __builtin_amdgcn_wmma_f32_16x16x32_f16(
                   false, av, false, bv3, (short)0, acc3, false, false);
    }

    // C layout: VGPR r, lanes 0-15 -> M=r, lanes 16-31 -> M=8+r; N = lane%16.
    float* cbase = Cout + (size_t)(tm * 16 + lhi * 8) * HC + lmod;
#pragma unroll
    for (int r = 0; r < 8; ++r) {
        float* cp = cbase + (size_t)r * HC;
        cp[(tn0 + 0) * 16] = acc0[r];
        cp[(tn0 + 1) * 16] = acc1[r];
        cp[(tn0 + 2) * 16] = acc2[r];
        cp[(tn0 + 3) * 16] = acc3[r];
    }
}

// --------------------------- attention pieces ------------------------------

// al_src[n,h] = sum_c h[n,h,c]*a_s[h,c] ; al_dst analogous. 1 thread per (n,h).
__global__ void attn_logits_kernel(const float* __restrict__ h,
                                   const float* __restrict__ a_s,
                                   const float* __restrict__ a_d,
                                   float* __restrict__ alsrc,
                                   float* __restrict__ aldst) {
    int i = blockIdx.x * blockDim.x + threadIdx.x;
    if (i >= N_NODES * H_HEADS) return;
    int n = i / H_HEADS;
    int hh = i % H_HEADS;
    const float4* hp = (const float4*)(h + (size_t)n * HC + hh * C_DIM);
    const float4* sp = (const float4*)(a_s + hh * C_DIM);
    const float4* dp = (const float4*)(a_d + hh * C_DIM);
    float ss = 0.0f, sd = 0.0f;
#pragma unroll
    for (int j = 0; j < C_DIM / 4; ++j) {
        float4 v = hp[j];
        float4 s = sp[j];
        float4 d = dp[j];
        ss += v.x * s.x + v.y * s.y + v.z * s.z + v.w * s.w;
        sd += v.x * d.x + v.y * d.y + v.z * d.z + v.w * d.w;
    }
    alsrc[i] = ss;
    aldst[i] = sd;
}

__device__ __forceinline__ void edge_nodes(const int* __restrict__ ei, int e,
                                           int& s, int& d) {
    if (e < N_EDGES) { s = ei[e]; d = ei[N_EDGES + e]; }
    else             { s = d = e - N_EDGES; }            // self loop
}

__device__ __forceinline__ float leaky(float x) {
    return x > 0.0f ? x : NEG_SLOPE * x;
}

// pass 1: per-dst segment max of leaky_relu(alsrc[s]+aldst[d])
__global__ void edge_max_kernel(const int* __restrict__ ei,
                                const float* __restrict__ alsrc,
                                const float* __restrict__ aldst,
                                float* __restrict__ m) {
    int e = blockIdx.x * blockDim.x + threadIdx.x;
    if (e >= N_EE) return;
    int s, d;
    edge_nodes(ei, e, s, d);
#pragma unroll
    for (int hh = 0; hh < H_HEADS; ++hh) {
        float al = leaky(alsrc[s * H_HEADS + hh] + aldst[d * H_HEADS + hh]);
        atomicMaxF(&m[d * H_HEADS + hh], al);
    }
}

// pass 2: denom[d,h] += exp(al - m[d,h])
__global__ void edge_expsum_kernel(const int* __restrict__ ei,
                                   const float* __restrict__ alsrc,
                                   const float* __restrict__ aldst,
                                   const float* __restrict__ m,
                                   float* __restrict__ denom) {
    int e = blockIdx.x * blockDim.x + threadIdx.x;
    if (e >= N_EE) return;
    int s, d;
    edge_nodes(ei, e, s, d);
#pragma unroll
    for (int hh = 0; hh < H_HEADS; ++hh) {
        float al = leaky(alsrc[s * H_HEADS + hh] + aldst[d * H_HEADS + hh]);
        atomicAdd(&denom[d * H_HEADS + hh], expf(al - m[d * H_HEADS + hh]));
    }
}

// pass 3: out[d, h*64+c] += coef * h[s, h*64+c]. One 256-thread block per edge.
__global__ __launch_bounds__(HC)
void edge_aggregate_kernel(const int* __restrict__ ei,
                           const float* __restrict__ alsrc,
                           const float* __restrict__ aldst,
                           const float* __restrict__ m,
                           const float* __restrict__ denom,
                           const float* __restrict__ hproj,
                           float* __restrict__ hout) {
    int e = blockIdx.x;
    int t = threadIdx.x;              // t = h*64 + c
    int s, d;
    edge_nodes(ei, e, s, d);
    int hh = t >> 6;
    float al = leaky(alsrc[s * H_HEADS + hh] + aldst[d * H_HEADS + hh]);
    float ev = expf(al - m[d * H_HEADS + hh]);
    float coef = ev / (denom[d * H_HEADS + hh] + 1e-16f);
    atomicAdd(&hout[(size_t)d * HC + t], coef * hproj[(size_t)s * HC + t]);
}

// bias + relu; also emit f16 copy for the next layer's WMMA input.
__global__ void bias_relu_kernel(float* __restrict__ h,
                                 const float* __restrict__ b,
                                 _Float16* __restrict__ h16) {
    int i = blockIdx.x * blockDim.x + threadIdx.x;
    if (i >= N_NODES * HC) return;
    float v = h[i] + b[i % HC];
    v = fmaxf(v, 0.0f);
    h[i] = v;
    h16[i] = (_Float16)v;
}

// ------------------------------- pooling -----------------------------------

__global__ void head_mean_kernel(const float* __restrict__ h,
                                 float* __restrict__ hm) {
    int i = blockIdx.x * blockDim.x + threadIdx.x;   // i = n*64 + c
    if (i >= N_NODES * C_DIM) return;
    int n = i / C_DIM;
    int c = i % C_DIM;
    const float* p = h + (size_t)n * HC + c;
    hm[i] = 0.25f * (p[0] + p[C_DIM] + p[2 * C_DIM] + p[3 * C_DIM]);
}

__global__ void pool_kernel(const float* __restrict__ hm,
                            const int* __restrict__ batch,
                            float* __restrict__ gmax,
                            float* __restrict__ gsum,
                            float* __restrict__ cnt) {
    int i = blockIdx.x * blockDim.x + threadIdx.x;   // i = n*64 + c
    if (i >= N_NODES * C_DIM) return;
    int n = i / C_DIM;
    int c = i % C_DIM;
    int g = batch[n];
    float v = hm[i];
    atomicMaxF(&gmax[g * C_DIM + c], v);
    atomicAdd(&gsum[g * C_DIM + c], v);
    if (c == 0) atomicAdd(&cnt[g], 1.0f);
}

__global__ void final_linear_kernel(const float* __restrict__ gmax,
                                    const float* __restrict__ gsum,
                                    const float* __restrict__ cnt,
                                    const float* __restrict__ Wout,
                                    const float* __restrict__ bout,
                                    float* __restrict__ out) {
    int i = blockIdx.x * blockDim.x + threadIdx.x;
    if (i >= N_B * N_OUT) return;
    int b = i / N_OUT;
    int o = i % N_OUT;
    float acc = bout[o];
    float invc = 1.0f / fmaxf(cnt[b], 1.0f);
#pragma unroll 4
    for (int j = 0; j < C_DIM; ++j) {
        float mx = gmax[b * C_DIM + j];
        if (!isfinite(mx)) mx = 0.0f;                 // empty-graph guard
        float s = gsum[b * C_DIM + j];
        acc += mx * Wout[j * N_OUT + o];
        acc += (s * invc) * Wout[(C_DIM + j) * N_OUT + o];
        acc += s * Wout[(2 * C_DIM + j) * N_OUT + o];
    }
    out[i] = acc;
}

// ------------------------------- driver ------------------------------------

static inline int cdiv(int a, int b) { return (a + b - 1) / b; }

extern "C" void kernel_launch(void* const* d_in, const int* in_sizes, int n_in,
                              void* d_out, int out_size, void* d_ws, size_t ws_size,
                              hipStream_t stream) {
    (void)in_sizes; (void)n_in; (void)out_size; (void)ws_size;

    const float* x        = (const float*)d_in[0];
    const int*   ei       = (const int*)d_in[1];
    const int*   batch    = (const int*)d_in[2];
    const float* W[3]     = {(const float*)d_in[3], (const float*)d_in[7],  (const float*)d_in[11]};
    const float* asrc[3]  = {(const float*)d_in[4], (const float*)d_in[8],  (const float*)d_in[12]};
    const float* adst[3]  = {(const float*)d_in[5], (const float*)d_in[9],  (const float*)d_in[13]};
    const float* bias[3]  = {(const float*)d_in[6], (const float*)d_in[10], (const float*)d_in[14]};
    const float* Wout     = (const float*)d_in[15];
    const float* bout     = (const float*)d_in[16];
    float* out            = (float*)d_out;

    // workspace layout (all offsets multiples of 256 bytes)
    char* ws = (char*)d_ws;
    size_t off = 0;
    auto alloc = [&](size_t bytes) {
        void* p = ws + off;
        off += (bytes + 255) & ~(size_t)255;
        return p;
    };
    _Float16* h16   = (_Float16*)alloc((size_t)N_NODES * HC * 2);  // gemm input
    _Float16* w16t  = (_Float16*)alloc((size_t)HC * HC * 2);       // Wt (f16)
    float*    hproj = (float*)alloc((size_t)N_NODES * HC * 4);     // x @ W
    float*    hout  = (float*)alloc((size_t)N_NODES * HC * 4);     // aggregated
    float*    alsrc = (float*)alloc((size_t)N_NODES * H_HEADS * 4);
    float*    aldst = (float*)alloc((size_t)N_NODES * H_HEADS * 4);
    float*    mbuf  = (float*)alloc((size_t)N_NODES * H_HEADS * 4);
    float*    dbuf  = (float*)alloc((size_t)N_NODES * H_HEADS * 4);
    float*    gmax  = (float*)alloc((size_t)N_B * C_DIM * 4);
    float*    gsum  = (float*)alloc((size_t)N_B * C_DIM * 4);
    float*    cnt   = (float*)alloc((size_t)N_B * 4);
    float*    hm    = hproj;  // reuse after last layer's gemm output is consumed

    const float NEG_INF = -INFINITY;
    const int TPB = 256;

    // layer 0 input -> f16
    cvt_f32_to_f16_kernel<<<cdiv(N_NODES * F_IN, TPB), TPB, 0, stream>>>(
        x, h16, N_NODES * F_IN);

    for (int l = 0; l < 3; ++l) {
        const int K = (l == 0) ? F_IN : HC;

        // weights -> transposed f16
        transpose_cvt_w_kernel<<<cdiv(K * HC, TPB), TPB, 0, stream>>>(
            W[l], w16t, K);

        // init per-layer buffers
        fill_f32_kernel<<<cdiv(N_NODES * H_HEADS, TPB), TPB, 0, stream>>>(
            mbuf, NEG_INF, N_NODES * H_HEADS);
        fill_f32_kernel<<<cdiv(N_NODES * H_HEADS, TPB), TPB, 0, stream>>>(
            dbuf, 0.0f, N_NODES * H_HEADS);
        fill_f32_kernel<<<cdiv(N_NODES * HC, TPB), TPB, 0, stream>>>(
            hout, 0.0f, N_NODES * HC);

        // projection: hproj = h16 @ W  (WMMA, f32 accumulate, 16x64 per wave)
        wmma_gemm_kernel<<<(N_NODES / 16) * (HC / 64), 32, 0, stream>>>(
            h16, w16t, hproj, K);

        // attention logits per node/head
        attn_logits_kernel<<<cdiv(N_NODES * H_HEADS, TPB), TPB, 0, stream>>>(
            hproj, asrc[l], adst[l], alsrc, aldst);

        // edge softmax (max, exp-sum) + weighted scatter-add
        edge_max_kernel<<<cdiv(N_EE, TPB), TPB, 0, stream>>>(
            ei, alsrc, aldst, mbuf);
        edge_expsum_kernel<<<cdiv(N_EE, TPB), TPB, 0, stream>>>(
            ei, alsrc, aldst, mbuf, dbuf);
        edge_aggregate_kernel<<<N_EE, HC, 0, stream>>>(
            ei, alsrc, aldst, mbuf, dbuf, hproj, hout);

        // bias + relu; produce next layer's f16 input
        bias_relu_kernel<<<cdiv(N_NODES * HC, TPB), TPB, 0, stream>>>(
            hout, bias[l], h16);
    }

    // mean over heads -> [N, 64] (reuses hproj storage)
    head_mean_kernel<<<cdiv(N_NODES * C_DIM, TPB), TPB, 0, stream>>>(hout, hm);

    // per-graph pooling
    fill_f32_kernel<<<cdiv(N_B * C_DIM, TPB), TPB, 0, stream>>>(
        gmax, NEG_INF, N_B * C_DIM);
    fill_f32_kernel<<<cdiv(N_B * C_DIM, TPB), TPB, 0, stream>>>(
        gsum, 0.0f, N_B * C_DIM);
    fill_f32_kernel<<<cdiv(N_B, TPB), TPB, 0, stream>>>(cnt, 0.0f, N_B);
    pool_kernel<<<cdiv(N_NODES * C_DIM, TPB), TPB, 0, stream>>>(
        hm, batch, gmax, gsum, cnt);

    // final linear head
    final_linear_kernel<<<cdiv(N_B * N_OUT, TPB), TPB, 0, stream>>>(
        gmax, gsum, cnt, Wout, bout, out);
}
